// YoloBboxLoss_65609920414062
// MI455X (gfx1250) — compile-verified, matrix operand order checked by reading, not run on hardware
//
#include <hip/hip_runtime.h>
#include <math.h>

typedef __attribute__((ext_vector_type(16))) _Float16 v16h;
typedef __attribute__((ext_vector_type(8)))  float    v8f;

#define Bc   16
#define Lc   100
#define NCc  80
#define NCH  85
#define OVc  0.5f

// anchors / stride, per level
__constant__ float c_anch[3][3][2] = {
  {{10.f/8.f, 13.f/8.f},   {16.f/8.f, 30.f/8.f},   {33.f/8.f, 23.f/8.f}},
  {{30.f/16.f, 61.f/16.f}, {62.f/16.f, 45.f/16.f}, {59.f/16.f, 119.f/16.f}},
  {{116.f/32.f, 90.f/32.f},{156.f/32.f,198.f/32.f},{373.f/32.f,326.f/32.f}},
};
__constant__ float c_offs[5][2] = {{0.f,0.f},{-OVc,0.f},{0.f,-OVc},{OVc,0.f},{0.f,OVc}};

__device__ __forceinline__ float softplusf(float x) {
  return fmaxf(x, 0.f) + log1pf(expf(-fabsf(x)));
}
__device__ __forceinline__ float sigmoidf(float x) {
  return 1.f / (1.f + expf(-x));
}
__device__ __forceinline__ float waveSum(float v) {
  #pragma unroll
  for (int off = 16; off > 0; off >>= 1) v += __shfl_xor(v, off, 32);
  return v;
}

// forward CIoU, xywh (center) boxes; alpha uses forward value (stop_gradient)
__device__ float ciouf(float b1x, float b1y, float b1w, float b1h,
                       float b2x, float b2y, float b2w, float b2h) {
  const float eps = 1e-9f;
  float b1x1 = b1x - 0.5f*b1w, b1x2 = b1x + 0.5f*b1w;
  float b1y1 = b1y - 0.5f*b1h, b1y2 = b1y + 0.5f*b1h;
  float b2x1 = b2x - 0.5f*b2w, b2x2 = b2x + 0.5f*b2w;
  float b2y1 = b2y - 0.5f*b2h, b2y2 = b2y + 0.5f*b2h;
  float iw = fmaxf(fminf(b1x2, b2x2) - fmaxf(b1x1, b2x1), 0.f);
  float ih = fmaxf(fminf(b1y2, b2y2) - fmaxf(b1y1, b2y1), 0.f);
  float inter = iw * ih;
  float w1 = b1x2 - b1x1, h1 = b1y2 - b1y1 + eps;
  float w2 = b2x2 - b2x1, h2 = b2y2 - b2y1 + eps;
  float uni = w1*h1 + w2*h2 - inter + eps;
  float iou = inter / uni;
  float cw = fmaxf(b1x2, b2x2) - fminf(b1x1, b2x1);
  float ch = fmaxf(b1y2, b2y2) - fminf(b1y1, b2y1);
  float c2 = cw*cw + ch*ch + eps;
  float dx = b2x1 + b2x2 - b1x1 - b1x2;
  float dy = b2y1 + b2y2 - b1y1 - b1y2;
  float rho2 = (dx*dx + dy*dy) * 0.25f;
  const float k = 4.f / (float)(M_PI * M_PI);
  float da = atanf(w2 / h2) - atanf(w1 / h1);
  float v = k * da * da;
  float alpha = v / (1.f + eps - iou + v);
  return iou - (rho2 / c2 + v * alpha);
}

__global__ void zero_ws_kernel(float* ws, int n) {
  int stride = gridDim.x * blockDim.x;
  for (int i = blockIdx.x * blockDim.x + threadIdx.x; i < n; i += stride) ws[i] = 0.f;
}

__global__ void nlabel_kernel(const float* __restrict__ ytrue, int* __restrict__ nlabel) {
  __shared__ int cnt;
  if (threadIdx.x == 0) cnt = 0;
  __syncthreads();
  int b = blockIdx.x;
  for (int l = threadIdx.x; l < Lc; l += blockDim.x) {
    const float* r = ytrue + ((size_t)b * Lc + l) * 5;
    float s = r[0] + r[1] + r[2] + r[3] + r[4];
    if (s > 0.f) atomicAdd(&cnt, 1);
  }
  __syncthreads();
  if (threadIdx.x == 0) nlabel[b] = cnt;
}

// One wave = 16 tuples. Scalar path: CIoU + gt_obj scatter + (-pred[label]) cls term.
// WMMA path: masked softplus over 80 classes summed via 3x v_wmma_f32_16x16x32_f16
// against an all-ones B (batched 16-row reduction in f32 accumulators).
__global__ __launch_bounds__(128)
void tuple_kernel(const float* __restrict__ pred, const float* __restrict__ ytrue,
                  const int* __restrict__ nlabel, float* __restrict__ gtobj,
                  float* __restrict__ acc, int li, float inv_s, int H, int W) {
  const int lane = threadIdx.x & 31;
  const int row  = lane & 15;
  const int half = lane >> 4;
  const int wave = blockIdx.x * (blockDim.x >> 5) + (threadIdx.x >> 5);
  const int tupleIdx = wave * 16 + row;            // 24000 tuples = 1500 full waves

  int g = tupleIdx / 15, rem = tupleIdx % 15;
  int a = rem / 5, o = rem % 5;
  int b = g / Lc, l = g % Lc;

  const float* yt = ytrue + (size_t)g * 5;
  float tw = yt[2], th = yt[3];
  float sx = (yt[0] + 0.5f * tw) * inv_s;
  float sy = (yt[1] + 0.5f * th) * inv_s;
  float sw = tw * inv_s, sh = th * inv_s;
  int label = (int)yt[4];
  label = min(max(label, 0), NCc - 1);

  bool valid = l < nlabel[b];
  float aw = c_anch[li][a][0], ah = c_anch[li][a][1];
  float rw = sw / aw, rh = sh / ah;
  float maxr = fmaxf(fmaxf(rw, 1.f / rw), fmaxf(rh, 1.f / rh));
  bool match = (maxr < 4.f) && valid;

  float fx = sx - floorf(sx), fy = sy - floorf(sy);
  bool cond;
  switch (o) {
    default: cond = true; break;
    case 1:  cond = (sx > 1.f) && (fx < OVc); break;
    case 2:  cond = (sy > 1.f) && (fy < OVc); break;
    case 3:  cond = (sx < (float)(W - 1)) && (fx > 1.f - OVc); break;
    case 4:  cond = (sy < (float)(H - 1)) && (fy > 1.f - OVc); break;
  }
  bool cmask = match && cond;
  float cm = cmask ? 1.f : 0.f;

  int gx = (int)(sx + c_offs[o][0]);
  int gy = (int)(sy + c_offs[o][1]);
  int gxc = min(max(gx, 0), W - 1);
  int gyc = min(max(gy, 0), H - 1);
  const int HW = H * W;
  const float* ptr = pred + (size_t)(b * 3 + a) * NCH * HW + (size_t)gyc * W + gxc;

  // ---- scalar per-tuple work (row owners: lanes 0..15) ----
  float box_c = 0.f, nval_c = 0.f, clssub_c = 0.f;
  if (half == 0) {
    float p0 = ptr[0], p1 = ptr[HW], p2 = ptr[2 * HW], p3 = ptr[3 * HW];
    float px = sigmoidf(p0) + (float)gx;
    float py = sigmoidf(p1) + (float)gy;
    float pw = expf(p2) * aw;
    float ph = expf(p3) * ah;
    float ci = ciouf(px, py, pw, ph, sx, sy, sw, sh);
    box_c    = (1.f - ci) * cm;
    nval_c   = cm;
    clssub_c = -ptr[(size_t)(5 + label) * HW] * cm;   // -z_c * pred_c term
    if (cmask && gx >= 0 && gx < W && gy >= 0 && gy < H)
      gtobj[((size_t)(b * 3 + a) * H + gy) * W + gx] = fmaxf(ci, 0.f);
  }
  float rbx = waveSum(box_c), rnv = waveSum(nval_c), rcs = waveSum(clssub_c);
  if (lane == 0) {
    atomicAdd(&acc[li * 4 + 0], rbx);
    atomicAdd(&acc[li * 4 + 1], rnv);
    atomicAdd(&acc[li * 4 + 2], rcs);
  }

  // ---- WMMA: sum_c softplus(pred_c) over 80 classes, 16 tuples per wave ----
  v16h ones;
  #pragma unroll
  for (int e = 0; e < 16; ++e) ones[e] = (_Float16)1.0f;
  v8f cacc = {0.f, 0.f, 0.f, 0.f, 0.f, 0.f, 0.f, 0.f};
  #pragma unroll
  for (int chunk = 0; chunk < 3; ++chunk) {
    v16h amat;
    #pragma unroll
    for (int e = 0; e < 16; ++e) {
      int k = (e < 8 ? e : e + 8) + half * 8;   // 16-bit A 16x32 lane layout
      int cidx = chunk * 32 + k;
      float v = 0.f;
      if (cidx < NCc) {                          // uniform per unrolled iter
        float pc = ptr[(size_t)(5 + cidx) * HW]; // always-valid clamped address
        v = cmask ? softplusf(pc) : 0.f;         // mask rows instead of EXEC
      }
      amat[e] = (_Float16)v;
    }
    cacc = __builtin_amdgcn_wmma_f32_16x16x32_f16(
        false, amat, false, ones, (short)0, cacc, false, false);
  }
  if (row == 0) {  // lane 0 holds rows 0..7 at N=0; lane 16 holds rows 8..15
    float s = cacc[0] + cacc[1] + cacc[2] + cacc[3] +
              cacc[4] + cacc[5] + cacc[6] + cacc[7];
    atomicAdd(&acc[li * 4 + 2], s);
  }
}

// Dense obj BCE: sum softplus(p4) - gt_obj * p4 over B*3*H*W cells (channel 4 only)
__global__ void obj_kernel(const float* __restrict__ pred, const float* __restrict__ gtobj,
                           float* __restrict__ acc, int accIdx, int HW, int total) {
  float sum = 0.f;
  int stride = gridDim.x * blockDim.x;
  for (int i = blockIdx.x * blockDim.x + threadIdx.x; i < total; i += stride) {
    int ba = i / HW, yx = i - ba * HW;
    float x = pred[((size_t)ba * NCH + 4) * HW + yx];
    float t = gtobj[i];
    sum += softplusf(x) - t * x;
  }
  sum = waveSum(sum);
  if ((threadIdx.x & 31) == 0) atomicAdd(&acc[accIdx], sum);
}

__global__ void finalize_kernel(const float* __restrict__ acc, float* __restrict__ out) {
  if (blockIdx.x == 0 && threadIdx.x == 0) {
    const float cellcnt[3] = {16.f * 3.f * 6400.f, 16.f * 3.f * 1600.f, 16.f * 3.f * 400.f};
    float lb = 0.f, lo = 0.f, lc = 0.f;
    for (int li = 0; li < 3; ++li) {
      float nval  = acc[li * 4 + 1];
      float denom = fmaxf(nval, 1.f);
      if (nval > 0.f) {
        lb += acc[li * 4 + 0] / denom;
        lc += acc[li * 4 + 2] / (denom * (float)NCc);
      }
      lo += acc[li * 4 + 3] / cellcnt[li];
    }
    lb *= 0.05f; lo *= 1.0f; lc *= 0.58f;
    out[0] = lb + lo + lc;
    out[1] = lb;
    out[2] = lo;
    out[3] = lc;
  }
}

extern "C" void kernel_launch(void* const* d_in, const int* in_sizes, int n_in,
                              void* d_out, int out_size, void* d_ws, size_t ws_size,
                              hipStream_t stream) {
  const float* yp0 = (const float*)d_in[0];   // (16, 255, 80, 80)
  const float* yp1 = (const float*)d_in[1];   // (16, 255, 40, 40)
  const float* yp2 = (const float*)d_in[2];   // (16, 255, 20, 20)
  const float* yt  = (const float*)d_in[3];   // (16, 100, 5)

  float* wsf    = (float*)d_ws;
  float* acc    = wsf;                 // 12 accumulators (box, nval, cls, obj) x 3 levels
  int*   nlab   = (int*)(wsf + 16);    // 16 per-batch label counts
  float* gtobj  = wsf + 64;            // dense gt_obj: 307200 + 76800 + 19200 floats
  const int zeroN = 64 + 403200;

  zero_ws_kernel<<<512, 256, 0, stream>>>(wsf, zeroN);
  nlabel_kernel<<<Bc, 128, 0, stream>>>(yt, nlab);

  // 24000 tuples per level = 1500 waves of 16 tuples = 375 blocks of 128 threads
  tuple_kernel<<<375, 128, 0, stream>>>(yp0, yt, nlab, gtobj,          acc, 0, 1.f/8.f,  80, 80);
  tuple_kernel<<<375, 128, 0, stream>>>(yp1, yt, nlab, gtobj + 307200, acc, 1, 1.f/16.f, 40, 40);
  tuple_kernel<<<375, 128, 0, stream>>>(yp2, yt, nlab, gtobj + 384000, acc, 2, 1.f/32.f, 20, 20);

  obj_kernel<<<480, 256, 0, stream>>>(yp0, gtobj,          acc, 0 * 4 + 3, 6400, 16 * 3 * 6400);
  obj_kernel<<<240, 256, 0, stream>>>(yp1, gtobj + 307200, acc, 1 * 4 + 3, 1600, 16 * 3 * 1600);
  obj_kernel<<<120, 256, 0, stream>>>(yp2, gtobj + 384000, acc, 2 * 4 + 3, 400,  16 * 3 * 400);

  finalize_kernel<<<1, 32, 0, stream>>>(acc, (float*)d_out);
}
